// BipartiteGraphConvolution_63874753626721
// MI455X (gfx1250) — matrix operand order repaired and checked by reading, not compiled, not run on hardware
//
#include <hip/hip_runtime.h>
#include <hip/hip_bf16.h>

typedef __attribute__((ext_vector_type(16))) _Float16 v16h;
typedef __attribute__((ext_vector_type(8)))  _Float16 v8h;
typedef __attribute__((ext_vector_type(8)))  float    v8f;

#define EMB 64
#define STR 72   // LDS row stride in halves (144 B: 16B-aligned, conflict-skewed)

// ---------------- Kernel 1: sum of edge weights ----------------
__global__ void bgc_sum_w(const float* __restrict__ w, int m, float* __restrict__ total) {
    float s = 0.0f;
    for (int i = blockIdx.x * blockDim.x + threadIdx.x; i < m; i += gridDim.x * blockDim.x)
        s += w[i];
    #pragma unroll
    for (int off = 16; off > 0; off >>= 1)
        s += __shfl_down(s, off, 32);
    if ((threadIdx.x & 31) == 0)
        atomicAdd(total, s);
}

// ---------------- Kernel 2: edge gather + scatter-add (unscaled) ----------------
// item = edge*4 + quarter; each thread handles 16 features of one edge.
__global__ void bgc_scatter(const int* __restrict__ ei, const float* __restrict__ w,
                            const float* __restrict__ left, float* __restrict__ conv,
                            long long items) {
    long long t = (long long)blockIdx.x * blockDim.x + threadIdx.x;
    if (t >= items) return;
    int e = (int)(t >> 2);
    int q = (int)(t & 3);
    int row = ei[2 * e];       // destination right node
    int col = ei[2 * e + 1];   // source left node
    float wt = w[e];
    const float4* src = (const float4*)(left + (size_t)col * EMB + q * 16);
    float* dst = conv + (size_t)row * EMB + q * 16;
    #pragma unroll
    for (int j = 0; j < 4; ++j) {
        float4 v = src[j];
        atomicAdd(dst + 4 * j + 0, v.x * wt);
        atomicAdd(dst + 4 * j + 1, v.y * wt);
        atomicAdd(dst + 4 * j + 2, v.z * wt);
        atomicAdd(dst + 4 * j + 3, v.w * wt);
    }
}

// ---------------- WMMA fragment helpers (CDNA5 16x16x32 f16 layouts) ----------------
// A (16x32, MxK): lane L -> M=L%16; half h -> K = 32*chunk + 16*(h/8) + 8*(L/16) + h%8
__device__ inline v16h loadA(const _Float16* __restrict__ tile, int lane, int chunk) {
    int m = lane & 15, g = lane >> 4;
    const v8h lo = *(const v8h*)(tile + m * STR + chunk * 32 + g * 8);
    const v8h hi = *(const v8h*)(tile + m * STR + chunk * 32 + 16 + g * 8);
    v16h a;
    #pragma unroll
    for (int j = 0; j < 8; ++j) { a[j] = lo[j]; a[8 + j] = hi[j]; }
    return a;
}

// B (32x16, KxN) with B[k][n] = W[n][k]: lane L -> n = 16*t + L%16;
// half h -> k = 32*chunk + 16*(h/8) + 8*(L/16) + h%8
__device__ inline v16h loadB(const _Float16* __restrict__ sW, int lane, int t, int chunk) {
    int nrow = t * 16 + (lane & 15);
    int g = lane >> 4;
    const _Float16* p = sW + nrow * STR + chunk * 32 + g * 8;
    const v8h lo = *(const v8h*)(p);
    const v8h hi = *(const v8h*)(p + 16);
    v16h b;
    #pragma unroll
    for (int j = 0; j < 8; ++j) { b[j] = lo[j]; b[8 + j] = hi[j]; }
    return b;
}

// ---------------- Kernel 3: fused h-build + 2-layer MLP via WMMA ----------------
// 256 threads = 8 waves; each wave owns a 16-node tile (block covers 128 nodes).
__global__ void bgc_fused_mlp(const float* __restrict__ rf, const float* __restrict__ cvec,
                              const float* __restrict__ conv, const float* __restrict__ totalp,
                              const float* __restrict__ temp,
                              const float* __restrict__ W1, const float* __restrict__ b1,
                              const float* __restrict__ W2, const float* __restrict__ b2,
                              float* __restrict__ out, int n) {
    __shared__ __align__(16) _Float16 sW1[64 * STR];
    __shared__ __align__(16) _Float16 sW2[64 * STR];
    __shared__ __align__(16) _Float16 sh[8][16 * STR];

    int tid = threadIdx.x;

    // Convert weights to f16 in LDS once per block
    for (int i = tid; i < 64 * 64; i += 256) {
        int r = i >> 6, cc = i & 63;
        sW1[r * STR + cc] = (_Float16)W1[i];
        sW2[r * STR + cc] = (_Float16)W2[i];
    }
    __syncthreads();

    const float invTotal = 1.0f / fmaxf(totalp[0], 1.0f);
    const float t1 = temp[1];

    int wave = tid >> 5;
    int lane = tid & 31;
    int g = lane >> 4;
    int nodeBase = blockIdx.x * 128 + wave * 16;
    _Float16* tile = &sh[wave][0];

    // Phase 1: h0 = rf + t1*(c - conv/total), f32 -> f16 tile (16 rows x 64 cols)
    #pragma unroll
    for (int it = 0; it < 8; ++it) {
        int idx = it * 32 + lane;       // 0..255 over 16x16 float4 grid
        int r = idx >> 4;               // row 0..15
        int c4 = (idx & 15) * 4;        // col
        int node = nodeBase + r;
        int nc = node < n ? node : n - 1;
        float4 rv = *(const float4*)(rf   + (size_t)nc * EMB + c4);
        float4 cv = *(const float4*)(conv + (size_t)nc * EMB + c4);
        float cs = cvec[nc];
        v8h* dst2 = nullptr; (void)dst2;
        __attribute__((ext_vector_type(4))) _Float16 hv;
        hv[0] = (_Float16)(rv.x + t1 * (cs - cv.x * invTotal));
        hv[1] = (_Float16)(rv.y + t1 * (cs - cv.y * invTotal));
        hv[2] = (_Float16)(rv.z + t1 * (cs - cv.z * invTotal));
        hv[3] = (_Float16)(rv.w + t1 * (cs - cv.w * invTotal));
        *(__attribute__((ext_vector_type(4))) _Float16*)(tile + r * STR + c4) = hv;
    }
    __syncthreads();

    // Layer 1: relu(h0 @ W1^T + b1) -> back into tile as f16
    {
        v16h a0 = loadA(tile, lane, 0);
        v16h a1 = loadA(tile, lane, 1);
        v8f accs[4];
        #pragma unroll
        for (int t = 0; t < 4; ++t) {
            float bv = b1[t * 16 + (lane & 15)];
            v8f acc;
            #pragma unroll
            for (int j = 0; j < 8; ++j) acc[j] = bv;
            v16h bb0 = loadB(sW1, lane, t, 0);
            v16h bb1 = loadB(sW1, lane, t, 1);
            acc = __builtin_amdgcn_wmma_f32_16x16x32_f16(false, a0, false, bb0, (short)0, acc, false, false);
            acc = __builtin_amdgcn_wmma_f32_16x16x32_f16(false, a1, false, bb1, (short)0, acc, false, false);
            accs[t] = acc;
        }
        // relu + writeback (D layout: VGPR r -> M = r + 8*g, N = lane%16)
        #pragma unroll
        for (int t = 0; t < 4; ++t) {
            int colw = t * 16 + (lane & 15);
            #pragma unroll
            for (int r = 0; r < 8; ++r) {
                int M = r + 8 * g;
                tile[M * STR + colw] = (_Float16)fmaxf(accs[t][r], 0.0f);
            }
        }
    }
    __syncthreads();

    // Layer 2: h1 @ W2^T + b2 -> global out (f32)
    {
        v16h a0 = loadA(tile, lane, 0);
        v16h a1 = loadA(tile, lane, 1);
        #pragma unroll
        for (int t = 0; t < 4; ++t) {
            float bv = b2[t * 16 + (lane & 15)];
            v8f acc;
            #pragma unroll
            for (int j = 0; j < 8; ++j) acc[j] = bv;
            v16h bb0 = loadB(sW2, lane, t, 0);
            v16h bb1 = loadB(sW2, lane, t, 1);
            acc = __builtin_amdgcn_wmma_f32_16x16x32_f16(false, a0, false, bb0, (short)0, acc, false, false);
            acc = __builtin_amdgcn_wmma_f32_16x16x32_f16(false, a1, false, bb1, (short)0, acc, false, false);
            int colw = t * 16 + (lane & 15);
            #pragma unroll
            for (int r = 0; r < 8; ++r) {
                int node = nodeBase + r + 8 * g;
                if (node < n)
                    out[(size_t)node * EMB + colw] = acc[r];
            }
        }
    }
}

extern "C" void kernel_launch(void* const* d_in, const int* in_sizes, int n_in,
                              void* d_out, int out_size, void* d_ws, size_t ws_size,
                              hipStream_t stream) {
    const float* left  = (const float*)d_in[0];
    const int*   ei    = (const int*)d_in[2];
    const float* ew    = (const float*)d_in[3];
    const float* rf    = (const float*)d_in[4];
    const float* cvec  = (const float*)d_in[5];
    const float* temp  = (const float*)d_in[7];
    const float* W1    = (const float*)d_in[8];
    const float* b1    = (const float*)d_in[9];
    const float* W2    = (const float*)d_in[10];
    const float* b2    = (const float*)d_in[11];
    float* out = (float*)d_out;

    int nEdges = in_sizes[3];
    int nRight = in_sizes[4] / EMB;

    float* total = (float*)d_ws;
    float* conv  = (float*)d_ws + 64;   // 256-byte offset, then 100000*64 f32 accumulator

    hipMemsetAsync(d_ws, 0, (64 + (size_t)nRight * EMB) * sizeof(float), stream);

    bgc_sum_w<<<512, 256, 0, stream>>>(ew, nEdges, total);

    long long items = (long long)nEdges * 4;
    int sblocks = (int)((items + 255) / 256);
    bgc_scatter<<<sblocks, 256, 0, stream>>>(ei, ew, left, conv, items);

    int fblocks = (nRight + 127) / 128;
    bgc_fused_mlp<<<fblocks, 256, 0, stream>>>(rf, cvec, conv, total, temp,
                                               W1, b1, W2, b2, out, nRight);
}
